// MyEncoderLayer_7241314861351
// MI455X (gfx1250) — matrix operand order repaired and checked by reading, not compiled, not run on hardware
//
#include <hip/hip_runtime.h>
#include <math.h>

// ---------------------------------------------------------------------------
// MyEncoderLayer on gfx1250 (CDNA5): LN -> QKV proj -> masked attention -> out proj
// All GEMMs via v_wmma_f32_16x16x32_f16 (f16 in, f32 accumulate).
// Projection GEMM: 32x32 tile per wave (2x2 register blocking, 4 WMMAs/k-step).
// ---------------------------------------------------------------------------

typedef __attribute__((ext_vector_type(16))) _Float16 v16h;
typedef __attribute__((ext_vector_type(8)))  _Float16 v8h;
typedef __attribute__((ext_vector_type(8)))  float    v8f;

#define BB 2
#define SS 2048
#define EE 768
#define HH 12
#define DD 64
#define MM (BB * SS)          // 4096 tokens
#define QT (SS / 16)          // 128 query tiles per (b,h)

static __device__ __forceinline__ v8f wmma16(v16h a, v16h b, v8f c) {
  // D = A(16x32 f16) * B(32x16 f16) + C(16x16 f32)
  return __builtin_amdgcn_wmma_f32_16x16x32_f16(false, a, false, b, (short)0, c,
                                                false, false);
}

static __device__ __forceinline__ v16h pack2(v8h lo, v8h hi) {
  v16h r;
#pragma unroll
  for (int i = 0; i < 8; ++i) { r[i] = lo[i]; r[8 + i] = hi[i]; }
  return r;
}

// ---------------------------------------------------------------------------
// 1) LayerNorm: x[4096,768] f32 -> xn[4096,768] f16
// ---------------------------------------------------------------------------
__global__ void ln_kernel(const float* __restrict__ x,
                          const float* __restrict__ w,
                          const float* __restrict__ b,
                          _Float16* __restrict__ xn) {
  const int t = blockIdx.x;
  const int tid = threadIdx.x;            // 256 threads, 3 elems each
  const float* xr = x + (size_t)t * EE;
  __shared__ float red[256];

  float v0 = xr[tid], v1 = xr[tid + 256], v2 = xr[tid + 512];
  red[tid] = v0 + v1 + v2;
#pragma unroll
  for (int off = 128; off > 0; off >>= 1) {
    __syncthreads();
    if (tid < off) red[tid] += red[tid + off];
  }
  __syncthreads();
  const float mu = red[0] * (1.0f / EE);
  __syncthreads();

  const float d0 = v0 - mu, d1 = v1 - mu, d2 = v2 - mu;
  red[tid] = d0 * d0 + d1 * d1 + d2 * d2;
#pragma unroll
  for (int off = 128; off > 0; off >>= 1) {
    __syncthreads();
    if (tid < off) red[tid] += red[tid + off];
  }
  __syncthreads();
  const float rstd = rsqrtf(red[0] * (1.0f / EE) + 1e-5f);

  _Float16* o = xn + (size_t)t * EE;
  o[tid]       = (_Float16)(d0 * rstd * w[tid]       + b[tid]);
  o[tid + 256] = (_Float16)(d1 * rstd * w[tid + 256] + b[tid + 256]);
  o[tid + 512] = (_Float16)(d2 * rstd * w[tid + 512] + b[tid + 512]);
}

// ---------------------------------------------------------------------------
// 2) Weight prep: WT[n][k] = (f16) W[k][n]   (768x768)
// ---------------------------------------------------------------------------
__global__ void wtr_kernel(const float* __restrict__ w, _Float16* __restrict__ wt) {
  const int idx = blockIdx.x * 256 + threadIdx.x;   // over 768*768
  const int n = idx / EE;
  const int k = idx - n * EE;
  wt[idx] = (_Float16)w[(size_t)k * EE + n];
}

// ---------------------------------------------------------------------------
// 3) WMMA GEMM epilogue for one 16x16 C tile.
// MODE 0: Q  -> f16 [B,H,S,D], scaled by 1/sqrt(D)
// MODE 1: K  -> f16 [B,H,S,D]
// MODE 2: V  -> f16 [B,H,D,S] (transposed, contiguous tokens)
// MODE 3: out-> f32 [M,768]
// ---------------------------------------------------------------------------
template <int MODE>
static __device__ __forceinline__ void epilogue(v8f acc, int m0, int n0, int lr, int hi,
                                                const float* __restrict__ bias,
                                                _Float16* __restrict__ outh,
                                                float* __restrict__ outf) {
  const int n = n0 + lr;
  const float bv = bias[n];

  if (MODE == 3) {
#pragma unroll
    for (int r = 0; r < 8; ++r) {
      const int m = m0 + r + 8 * hi;            // C layout: row = r + 8*half
      outf[(size_t)m * EE + n] = acc[r] + bv;
    }
  } else {
    const int hh = n >> 6;                       // head
    const int dd = n & 63;                       // dim within head
    const int bi = m0 / SS;                      // batch (tile never crosses)
    const int sbase = m0 - bi * SS;
    if (MODE == 2) {
      v8h o;
#pragma unroll
      for (int r = 0; r < 8; ++r) o[r] = (_Float16)(acc[r] + bv);
      _Float16* p = outh + (((size_t)(bi * HH + hh) * DD + dd) * SS + sbase + 8 * hi);
      *(v8h*)p = o;                              // contiguous tokens
    } else {
      const float scl = (MODE == 0) ? 0.125f : 1.0f;   // 1/sqrt(64) folded into Q
#pragma unroll
      for (int r = 0; r < 8; ++r) {
        const int s = sbase + r + 8 * hi;
        outh[((size_t)(bi * HH + hh) * SS + s) * DD + dd] = (_Float16)((acc[r] + bv) * scl);
      }
    }
  }
}

// 32x32 tile per wave: A = xn/H (f16 [M,768] row-major), B = WT (f16 [N,768]).
template <int MODE>
__global__ void gemm32(const _Float16* __restrict__ A,
                       const _Float16* __restrict__ WT,
                       const float* __restrict__ bias,
                       _Float16* __restrict__ outh,
                       float* __restrict__ outf) {
  const int lane = (int)(threadIdx.x & 31);
  const int lr = lane & 15;
  const int hi = lane >> 4;
  const int n0 = blockIdx.x * 32;
  const int m0 = blockIdx.y * 32;

  // A operand: row = m0+lr (+16); halves i<8: k=kk+8*hi+i, i>=8: k=kk+16+8*hi+(i-8)
  const _Float16* ap0 = A + (size_t)(m0 + lr) * EE + 8 * hi;
  const _Float16* ap1 = ap0 + (size_t)16 * EE;
  // B operand: col = n0+lr (+16); halves i<8: k=kk+8*hi+i, i>=8: k=kk+16+8*hi+(i-8)
  const _Float16* bp0 = WT + (size_t)(n0 + lr) * EE + 8 * hi;
  const _Float16* bp1 = bp0 + (size_t)16 * EE;

  const v8f z = {0.f, 0.f, 0.f, 0.f, 0.f, 0.f, 0.f, 0.f};
  v8f acc00 = z, acc01 = z, acc10 = z, acc11 = z;

#pragma unroll 2
  for (int kk = 0; kk < EE; kk += 32) {
    __builtin_prefetch(ap0 + kk + 256, 0, 1);    // global_prefetch_b8
    __builtin_prefetch(bp0 + kk + 256, 0, 1);
    const v16h a0 = pack2(*(const v8h*)(ap0 + kk), *(const v8h*)(ap0 + kk + 16));
    const v16h a1 = pack2(*(const v8h*)(ap1 + kk), *(const v8h*)(ap1 + kk + 16));
    const v16h b0 = pack2(*(const v8h*)(bp0 + kk), *(const v8h*)(bp0 + kk + 16));
    const v16h b1 = pack2(*(const v8h*)(bp1 + kk), *(const v8h*)(bp1 + kk + 16));
    acc00 = wmma16(a0, b0, acc00);
    acc01 = wmma16(a0, b1, acc01);
    acc10 = wmma16(a1, b0, acc10);
    acc11 = wmma16(a1, b1, acc11);
  }

  epilogue<MODE>(acc00, m0,      n0,      lr, hi, bias, outh, outf);
  epilogue<MODE>(acc01, m0,      n0 + 16, lr, hi, bias, outh, outf);
  epilogue<MODE>(acc10, m0 + 16, n0,      lr, hi, bias, outh, outf);
  epilogue<MODE>(acc11, m0 + 16, n0 + 16, lr, hi, bias, outh, outf);
}

// ---------------------------------------------------------------------------
// 4) Attention: one wave per (b, h, 16-query tile). Flash-style over key chunks
// of 32. Scores computed transposed (A=K tile, B=Q tile) so C holds S^T with
// lane = query -> per-lane online softmax, one shfl_xor(16) to merge halves.
// Mask: key < query => -inf (attends to self + future only).
// ---------------------------------------------------------------------------
__global__ void attn_kernel(const _Float16* __restrict__ Q,   // [B,H,S,D], pre-scaled
                            const _Float16* __restrict__ K,   // [B,H,S,D]
                            const _Float16* __restrict__ Vt,  // [B,H,D,S]
                            _Float16* __restrict__ Hout) {    // [M,768] f16
  const int lane = (int)(threadIdx.x & 31);
  const int lr = lane & 15;
  const int hi = lane >> 4;

  int idx = blockIdx.x;
  const int qt = idx & (QT - 1); idx >>= 7;
  const int h = idx % HH;
  const int b = idx / HH;
  const int q0 = qt * 16;

  // Q as B operand (lane = query, d contiguous), both 32-d k-steps
  const _Float16* qb = Q + (((size_t)(b * HH + h) * SS) + q0 + lr) * DD;
  const v16h bq0 = pack2(*(const v8h*)(qb + 8 * hi), *(const v8h*)(qb + 16 + 8 * hi));        // d = 8*hi + i, 16 + 8*hi + (i-8)
  const v16h bq1 = pack2(*(const v8h*)(qb + 32 + 8 * hi), *(const v8h*)(qb + 48 + 8 * hi));   // d = 32 + 8*hi + i, 48 + 8*hi + (i-8)

  const _Float16* kbase = K + ((size_t)(b * HH + h) * SS) * DD;
  const _Float16* vbase = Vt + ((size_t)(b * HH + h) * DD) * SS;

  float m_run = -__builtin_inff();
  float l_run = 0.f;
  v8f o[4];
#pragma unroll
  for (int dt = 0; dt < 4; ++dt)
    o[dt] = (v8f){0.f, 0.f, 0.f, 0.f, 0.f, 0.f, 0.f, 0.f};

  const int kstart = q0 & ~31;
  for (int kc = kstart; kc < SS; kc += 32) {
    // ---- scores: two 16-key subtiles, each 2 WMMAs over d ----
    const _Float16* kp0 = kbase + (size_t)(kc + lr) * DD + 8 * hi;
    const _Float16* kp1 = kbase + (size_t)(kc + 16 + lr) * DD + 8 * hi;

    v8f s0 = {0.f, 0.f, 0.f, 0.f, 0.f, 0.f, 0.f, 0.f};
    v8f s1 = {0.f, 0.f, 0.f, 0.f, 0.f, 0.f, 0.f, 0.f};
    s0 = wmma16(pack2(*(const v8h*)(kp0),      *(const v8h*)(kp0 + 16)), bq0, s0);
    s0 = wmma16(pack2(*(const v8h*)(kp0 + 32), *(const v8h*)(kp0 + 48)), bq1, s0);
    s1 = wmma16(pack2(*(const v8h*)(kp1),      *(const v8h*)(kp1 + 16)), bq0, s1);
    s1 = wmma16(pack2(*(const v8h*)(kp1 + 32), *(const v8h*)(kp1 + 48)), bq1, s1);

    // ---- mask (only first chunk can straddle the diagonal) ----
    if (kc < q0 + 16) {
      const int q = q0 + lr;
#pragma unroll
      for (int r = 0; r < 8; ++r) {
        if (kc + r + 8 * hi < q)      s0[r] = -__builtin_inff();
        if (kc + 16 + r + 8 * hi < q) s1[r] = -__builtin_inff();
      }
    }

    // ---- online softmax (per lane = per query; merge halves via shfl 16) ----
    float cm = s0[0];
#pragma unroll
    for (int r = 0; r < 8; ++r) { cm = fmaxf(cm, s0[r]); cm = fmaxf(cm, s1[r]); }
    cm = fmaxf(cm, __shfl_xor(cm, 16));
    const float nm = fmaxf(m_run, cm);
    const float alpha = __expf(m_run - nm);
    m_run = nm;

    float p0[8], p1[8], psum = 0.f;
#pragma unroll
    for (int r = 0; r < 8; ++r) {
      p0[r] = __expf(s0[r] - nm);
      p1[r] = __expf(s1[r] - nm);
      psum += p0[r] + p1[r];
    }
    l_run = l_run * alpha + psum;
#pragma unroll
    for (int dt = 0; dt < 4; ++dt)
#pragma unroll
      for (int r = 0; r < 8; ++r) o[dt][r] *= alpha;

    // ---- repack P^T (C layout) into B operand (lane = query, k = key) ----
    v16h bp;
#pragma unroll
    for (int i = 0; i < 8; ++i) {
      const float w0 = __shfl_xor(p1[i], 16);   // keys 16..23 for upper half
      const float w1 = __shfl_xor(p0[i], 16);   // keys  8..15 for lower half
      bp[i]     = (_Float16)((hi == 0) ? p0[i] : w0);
      bp[8 + i] = (_Float16)((hi == 0) ? w1   : p1[i]);
    }

    // ---- O^T += V^T(16d x 32k) * P^T(32k x 16q), 4 d-tiles ----
#pragma unroll
    for (int dt = 0; dt < 4; ++dt) {
      const _Float16* vp = vbase + (size_t)(dt * 16 + lr) * SS + kc + 16 * hi;
      o[dt] = wmma16(*(const v16h*)(vp), bp, o[dt]);
    }
  }

  // ---- finalize: divide by total sum, store H (f16, [token, h*64 + d]) ----
  const float lt = l_run + __shfl_xor(l_run, 16);
  const float inv = 1.0f / lt;
  const int token = b * SS + q0 + lr;
  _Float16* hp = Hout + (size_t)token * EE + h * 64;
#pragma unroll
  for (int dt = 0; dt < 4; ++dt) {
    v8h o8;
#pragma unroll
    for (int r = 0; r < 8; ++r) o8[r] = (_Float16)(o[dt][r] * inv);
    *(v8h*)(hp + dt * 16 + 8 * hi) = o8;        // d = dt*16 + r + 8*hi contiguous in r
  }
}

// ---------------------------------------------------------------------------
// Launch
// ---------------------------------------------------------------------------
extern "C" void kernel_launch(void* const* d_in, const int* in_sizes, int n_in,
                              void* d_out, int out_size, void* d_ws, size_t ws_size,
                              hipStream_t stream) {
  (void)in_sizes; (void)n_in; (void)out_size; (void)ws_size;
  const float* x    = (const float*)d_in[0];
  const float* ln_w = (const float*)d_in[1];
  const float* ln_b = (const float*)d_in[2];
  const float* wq   = (const float*)d_in[3];
  const float* bq   = (const float*)d_in[4];
  const float* wk   = (const float*)d_in[5];
  const float* bk   = (const float*)d_in[6];
  const float* wv   = (const float*)d_in[7];
  const float* bv   = (const float*)d_in[8];
  const float* wo   = (const float*)d_in[9];
  const float* bo   = (const float*)d_in[10];
  float* out = (float*)d_out;

  char* ws = (char*)d_ws;
  const size_t XN_B = (size_t)MM * EE * 2;        // 6,291,456
  const size_t WT_B = (size_t)EE * EE * 2;        // 1,179,648
  _Float16* xn  = (_Float16*)(ws);
  _Float16* wqT = (_Float16*)(ws + XN_B);
  _Float16* wkT = wqT + (size_t)EE * EE;
  _Float16* wvT = wkT + (size_t)EE * EE;
  _Float16* woT = wvT + (size_t)EE * EE;
  _Float16* Qb  = (_Float16*)(ws + XN_B + 4 * WT_B);
  _Float16* Kb  = Qb + (size_t)MM * EE;
  _Float16* Vtb = Kb + (size_t)MM * EE;
  _Float16* Hb  = Vtb + (size_t)MM * EE;

  ln_kernel<<<MM, 256, 0, stream>>>(x, ln_w, ln_b, xn);

  const int wgrid = (EE * EE) / 256;
  wtr_kernel<<<wgrid, 256, 0, stream>>>(wq, wqT);
  wtr_kernel<<<wgrid, 256, 0, stream>>>(wk, wkT);
  wtr_kernel<<<wgrid, 256, 0, stream>>>(wv, wvT);
  wtr_kernel<<<wgrid, 256, 0, stream>>>(wo, woT);

  dim3 g(EE / 32, MM / 32);                        // 24 x 128 tiles of 32x32
  gemm32<0><<<g, 32, 0, stream>>>(xn, wqT, bq, Qb, nullptr);
  gemm32<1><<<g, 32, 0, stream>>>(xn, wkT, bk, Kb, nullptr);
  gemm32<2><<<g, 32, 0, stream>>>(xn, wvT, bv, Vtb, nullptr);

  attn_kernel<<<BB * HH * QT, 32, 0, stream>>>(Qb, Kb, Vtb, Hb);

  gemm32<3><<<g, 32, 0, stream>>>(Hb, woT, bo, nullptr, out);
}
